// MultiHeadAttention_78950088835661
// MI455X (gfx1250) — matrix operand order repaired and checked by reading, not compiled, not run on hardware
//
#include <hip/hip_runtime.h>
#include <hip/hip_bf16.h>

typedef __attribute__((ext_vector_type(16))) _Float16 v16h;
typedef __attribute__((ext_vector_type(8)))  _Float16 v8h;
typedef __attribute__((ext_vector_type(8)))  float    v8f;

#define NEG_INF (-__builtin_inff())

// ---------------------------------------------------------------------------
// Fragment loaders for v_wmma_f32_16x16x32_f16 (wave32).
//
// A (16x32 f16, MxK): lane L: r = L&15 is the row, hi = L>>4
// elems e hold K = 16*hi + e (matching B)
// -> two contiguous 16B loads from a row-major A row.
//
// B (32x16 f16, KxN), same contiguous k order as A: lane L: n = L&15 is the
// column, elems e hold K = 16*hi + e -> one contiguous 32B load when B is
// stored column-major in K (i.e. W[N][K] row-major, which matches x @ W^T).
// ---------------------------------------------------------------------------
__device__ inline v16h load_a_frag(const _Float16* __restrict__ row, int k0, int hi) {
  v8h c0 = *(const v8h*)(row + k0 + hi * 16);
  v8h c1 = *(const v8h*)(row + k0 + hi * 16 + 8);
  v16h r;
#pragma unroll
  for (int e = 0; e < 8; ++e) { r[e] = c0[e]; r[8 + e] = c1[e]; }
  return r;
}

__device__ inline v8f wmma_f16(v16h a, v16h b, v8f c) {
  return __builtin_amdgcn_wmma_f32_16x16x32_f16(false, a, false, b, (short)0, c,
                                                false, false);
}

// ---------------------------------------------------------------------------
// f32 -> f16 convert
// ---------------------------------------------------------------------------
__global__ void f32_to_f16_k(const float* __restrict__ src, _Float16* __restrict__ dst,
                             int n) {
  int i = blockIdx.x * blockDim.x + threadIdx.x;
  if (i < n) dst[i] = (_Float16)src[i];
}

// ---------------------------------------------------------------------------
// QKV projection: [4096,512] x W[1536][512] (x @ W^T). Each wave computes a
// 16x64 tile: one A fragment feeds 4 WMMAs per 32-wide k-step (A reuse 4x).
// Ping-pong software pipeline (unroll x2): stage A consumes fragment set A
// while set B loads, stage B consumes set B while set A reloads. Explicit
// two-buffer form avoids rotation copies and keeps loads overlapping WMMAs
// (partial s_wait_loadcnt). Final stage-B prefetch wraps (&511) harmlessly.
// Epilogue scatters per the reference's per-head [q|k|v] channel split
// (head = c/192, part = (c%192)/64, d = c%64) into q[B,H,S,64], k[B,H,S,64],
// vT[B,H,64,S] (f16).
// ---------------------------------------------------------------------------
__global__ __launch_bounds__(256) void gemm_qkv_k(
    const _Float16* __restrict__ xh, const _Float16* __restrict__ wh,
    const float* __restrict__ bias, _Float16* __restrict__ qb,
    _Float16* __restrict__ kb, _Float16* __restrict__ vT) {
  const int warp = threadIdx.x >> 5, lane = threadIdx.x & 31;
  const int wid = blockIdx.x * 8 + warp;          // 6144 waves = 256 x 24 tiles
  const int tm = wid & 255, tn = wid >> 8;        // tn in 0..23
  const int m0 = tm << 4, n0 = tn << 6;           // 16 x 64 tile
  const int n = lane & 15, hi = lane >> 4;

  const _Float16* arow = xh + (size_t)(m0 + n) * 512;
  const _Float16* brow0 = wh + (size_t)(n0 + n) * 512;

  v8f acc[4];
#pragma unroll
  for (int t = 0; t < 4; ++t)
#pragma unroll
    for (int i = 0; i < 8; ++i) acc[t][i] = 0.0f;

  // prologue: fragment set A <- k=0
  v16h aA = load_a_frag(arow, 0, hi);
  v16h bA[4];
#pragma unroll
  for (int t = 0; t < 4; ++t)
    bA[t] = *(const v16h*)(brow0 + (size_t)(t * 16) * 512 + hi * 16);

  for (int k0 = 0; k0 < 512; k0 += 64) {
    // stage A: load set B (k0+32), consume set A (k0)
    const int k1 = k0 + 32;
    v16h aB = load_a_frag(arow, k1, hi);
    v16h bB[4];
#pragma unroll
    for (int t = 0; t < 4; ++t)
      bB[t] = *(const v16h*)(brow0 + (size_t)(t * 16) * 512 + k1 + hi * 16);
#pragma unroll
    for (int t = 0; t < 4; ++t) acc[t] = wmma_f16(aA, bA[t], acc[t]);

    // stage B: reload set A (k0+64, wrapped), consume set B (k0+32)
    const int k2 = (k0 + 64) & 511;
    aA = load_a_frag(arow, k2, hi);
#pragma unroll
    for (int t = 0; t < 4; ++t)
      bA[t] = *(const v16h*)(brow0 + (size_t)(t * 16) * 512 + k2 + hi * 16);
#pragma unroll
    for (int t = 0; t < 4; ++t) acc[t] = wmma_f16(aB, bB[t], acc[t]);
  }

#pragma unroll
  for (int t = 0; t < 4; ++t) {
    const int ng = n0 + t * 16 + n;
    const int head = ng / 192, rem = ng % 192, part = rem >> 6, d = rem & 63;
    const float bv = bias[ng];
#pragma unroll
    for (int i = 0; i < 8; ++i) {
      const int m = m0 + 8 * hi + i;              // row = (b, s)
      const int bidx = m >> 11, s = m & 2047;
      const size_t bh = (size_t)bidx * 8 + head;
      const _Float16 hv = (_Float16)(acc[t][i] + bv);
      if (part == 0)      qb[(bh * 2048 + s) * 64 + d] = hv;
      else if (part == 1) kb[(bh * 2048 + s) * 64 + d] = hv;
      else                vT[(bh * 64 + d) * 2048 + s] = hv;
    }
  }
}

// ---------------------------------------------------------------------------
// Banded attention, online softmax. One wave per (b, h, 16-query tile).
// Band |i-j| <= 128 aligns to 16 -> exactly 17 key tiles, edge tiles clipped.
// ---------------------------------------------------------------------------
__global__ __launch_bounds__(256) void attn_k(
    const _Float16* __restrict__ qb, const _Float16* __restrict__ kb,
    const _Float16* __restrict__ vT, const int* __restrict__ pm,
    _Float16* __restrict__ valsh) {
  __shared__ _Float16 ldsP[8][16][16];            // per-wave 16x16 P tile
  const int warp = threadIdx.x >> 5, lane = threadIdx.x & 31;
  const int wid = blockIdx.x * 8 + warp;          // 2048 waves
  const int qt = wid & 127, bh = wid >> 7;
  const int h = bh & 7, b = bh >> 3;
  const int q0 = qt << 4;
  const int n = lane & 15, hi = lane >> 4;

  const _Float16* qrow = qb + ((size_t)bh * 2048 + q0 + n) * 64;
  const _Float16* kbh  = kb + (size_t)bh * 2048 * 64;
  const _Float16* vbh  = vT + (size_t)bh * 64 * 2048;

  // Q A-fragments for both 32-wide K steps of HD=64
  const v16h a0 = load_a_frag(qrow, 0, hi);
  const v16h a1 = load_a_frag(qrow, 32, hi);

  float mrow[8], lrow[8];
  v8f O[4];
#pragma unroll
  for (int i = 0; i < 8; ++i) { mrow[i] = NEG_INF; lrow[i] = 0.0f; }
#pragma unroll
  for (int cc = 0; cc < 4; ++cc)
#pragma unroll
    for (int i = 0; i < 8; ++i) O[cc][i] = 0.0f;

  int qpad[8];
#pragma unroll
  for (int i = 0; i < 8; ++i) qpad[i] = pm[b * 2048 + q0 + 8 * hi + i];

  for (int kt = 0; kt < 17; ++kt) {
    const int j0 = q0 - 128 + kt * 16;
    if (j0 < 0 || j0 >= 2048) continue;           // per-wave LDS: divergence OK

    // ---- scores: S = Q Kt^T (2 WMMAs, both B fragments loaded first) ----
    const v16h b0 = *(const v16h*)(kbh + (size_t)(j0 + n) * 64 + hi * 16);
    const v16h b1 = *(const v16h*)(kbh + (size_t)(j0 + n) * 64 + 32 + hi * 16);
    v8f s;
#pragma unroll
    for (int i = 0; i < 8; ++i) s[i] = 0.0f;
    s = wmma_f16(a0, b0, s);
    s = wmma_f16(a1, b1, s);

    const int j = j0 + n;
    const bool kpad = (pm[b * 2048 + j] == 0);

    // ---- reference masking semantics + online softmax ----
#pragma unroll
    for (int i = 0; i < 8; ++i) {
      float sv = s[i] * 0.125f;                    // 1/sqrt(64)
      const int qi = q0 + 8 * hi + i;
      const int dj = qi - j;
      if (dj < -128 || dj > 128) sv = NEG_INF;     // off-band -> 0 -> -inf
      if (sv == 0.0f) sv = NEG_INF;                // exact zeros -> -inf
      if (kpad) sv = NEG_INF;                      // key padding

      float t = sv;                                // row max over 16 lanes
#pragma unroll
      for (int msk = 1; msk < 16; msk <<= 1) t = fmaxf(t, __shfl_xor(t, msk, 32));
      const float mn = fmaxf(mrow[i], t);

      float alpha, pe;
      if (mn == NEG_INF) { alpha = 1.0f; pe = 0.0f; }
      else {
        alpha = __expf(mrow[i] - mn);              // exp(-inf)=0 handles init
        pe = (sv == NEG_INF) ? 0.0f : __expf(sv - mn);
      }
      float ts = pe;                               // row sum over 16 lanes
#pragma unroll
      for (int msk = 1; msk < 16; msk <<= 1) ts += __shfl_xor(ts, msk, 32);
      lrow[i] = lrow[i] * alpha + ts;
      mrow[i] = mn;
#pragma unroll
      for (int cc = 0; cc < 4; ++cc) O[cc][i] *= alpha;

      ldsP[warp][8 * hi + i][n] = (_Float16)pe;    // C-layout -> LDS
    }

    asm volatile("s_wait_dscnt 0" ::: "memory");   // intra-wave LDS RAW fence

    // P as A-fragment: hi=0 elems 0..15 = keys 0..15, hi=1 (K 16..31) zero-padded
    const v16h pl = *(const v16h*)(&ldsP[warp][n][0]);
    v16h pa;
#pragma unroll
    for (int e = 0; e < 16; ++e) pa[e] = hi ? (_Float16)0.0f : pl[e];

    // ---- O += P V : load all 4 V fragments, then 4 WMMAs ----
    v16h vb[4];
#pragma unroll
    for (int cc = 0; cc < 4; ++cc) {
      if (hi == 0) {                               // K=0..15 (valid keys)
        vb[cc] = *(const v16h*)(vbh + (size_t)(cc * 16 + n) * 2048 + j0);
      } else {                                     // K=16..31 zero (P is zero)
#pragma unroll
        for (int e = 0; e < 16; ++e) vb[cc][e] = (_Float16)0.0f;
      }
    }
#pragma unroll
    for (int cc = 0; cc < 4; ++cc) O[cc] = wmma_f16(pa, vb[cc], O[cc]);
  }

  // ---- epilogue: normalize; masked rows (l==0) and padded queries -> 0 ----
#pragma unroll
  for (int i = 0; i < 8; ++i) {
    float rcp = (lrow[i] > 0.0f) ? (1.0f / lrow[i]) : 0.0f;
    if (qpad[i] == 0) rcp = 0.0f;
    const int srow = q0 + 8 * hi + i;
    const size_t base = ((size_t)b * 2048 + srow) * 512 + h * 64;
#pragma unroll
    for (int cc = 0; cc < 4; ++cc)
      valsh[base + cc * 16 + n] = (_Float16)(O[cc][i] * rcp);
  }
}

// ---------------------------------------------------------------------------
// Output projection: [4096,512] x w_o[512][512] (vals @ w_o^T) + b_o -> f32.
// 16x64 tile per wave, ping-pong pipelined like gemm_qkv_k.
// ---------------------------------------------------------------------------
__global__ __launch_bounds__(256) void gemm_out_k(
    const _Float16* __restrict__ valsh, const _Float16* __restrict__ woh,
    const float* __restrict__ bo, float* __restrict__ out) {
  const int warp = threadIdx.x >> 5, lane = threadIdx.x & 31;
  const int wid = blockIdx.x * 8 + warp;          // 2048 waves = 256 x 8 tiles
  const int tm = wid & 255, tn = wid >> 8;        // tn in 0..7
  const int m0 = tm << 4, n0 = tn << 6;
  const int n = lane & 15, hi = lane >> 4;

  const _Float16* arow = valsh + (size_t)(m0 + n) * 512;
  const _Float16* brow0 = woh + (size_t)(n0 + n) * 512;

  v8f acc[4];
#pragma unroll
  for (int t = 0; t < 4; ++t)
#pragma unroll
    for (int i = 0; i < 8; ++i) acc[t][i] = 0.0f;

  v16h aA = load_a_frag(arow, 0, hi);
  v16h bA[4];
#pragma unroll
  for (int t = 0; t < 4; ++t)
    bA[t] = *(const v16h*)(brow0 + (size_t)(t * 16) * 512 + hi * 16);

  for (int k0 = 0; k0 < 512; k0 += 64) {
    const int k1 = k0 + 32;
    v16h aB = load_a_frag(arow, k1, hi);
    v16h bB[4];
#pragma unroll
    for (int t = 0; t < 4; ++t)
      bB[t] = *(const v16h*)(brow0 + (size_t)(t * 16) * 512 + k1 + hi * 16);
#pragma unroll
    for (int t = 0; t < 4; ++t) acc[t] = wmma_f16(aA, bA[t], acc[t]);

    const int k2 = (k0 + 64) & 511;
    aA = load_a_frag(arow, k2, hi);
#pragma unroll
    for (int t = 0; t < 4; ++t)
      bA[t] = *(const v16h*)(brow0 + (size_t)(t * 16) * 512 + k2 + hi * 16);
#pragma unroll
    for (int t = 0; t < 4; ++t) acc[t] = wmma_f16(aB, bB[t], acc[t]);
  }

#pragma unroll
  for (int t = 0; t < 4; ++t) {
    const int ng = n0 + t * 16 + n;
    const float bv = bo[ng];
#pragma unroll
    for (int i = 0; i < 8; ++i) {
      const int m = m0 + 8 * hi + i;
      out[(size_t)m * 512 + ng] = acc[t][i] + bv;
    }
  }
}

// ---------------------------------------------------------------------------
// Launch. Inputs: x, padding_mask, w_qkv, b_qkv, w_o, b_o.
// Workspace layout (bytes, all 256B aligned):
//   xh      @ 0         : 4096*512   f16 = 4 MiB
//   wqkvh   @ 4194304   : 1536*512   f16 = 1.5 MiB
//   woh     @ 5767168   : 512*512    f16 = 0.5 MiB
//   qb      @ 6291456   : 16*2048*64 f16 = 4 MiB
//   kb      @ 10485760  : 4 MiB
//   vT      @ 14680064  : 4 MiB   ([B,H,64,S])
//   valsh   @ 18874368  : 4096*512 f16 = 4 MiB     (total ~22 MiB)
// ---------------------------------------------------------------------------
extern "C" void kernel_launch(void* const* d_in, const int* in_sizes, int n_in,
                              void* d_out, int out_size, void* d_ws, size_t ws_size,
                              hipStream_t stream) {
  const float* x      = (const float*)d_in[0];
  const int*   pmask  = (const int*)d_in[1];
  const float* w_qkv  = (const float*)d_in[2];
  const float* b_qkv  = (const float*)d_in[3];
  const float* w_o    = (const float*)d_in[4];
  const float* b_o    = (const float*)d_in[5];
  float* out = (float*)d_out;

  char* ws = (char*)d_ws;
  _Float16* xh    = (_Float16*)(ws + 0);
  _Float16* wqkvh = (_Float16*)(ws + 4194304);
  _Float16* woh   = (_Float16*)(ws + 5767168);
  _Float16* qb    = (_Float16*)(ws + 6291456);
  _Float16* kb    = (_Float16*)(ws + 10485760);
  _Float16* vT    = (_Float16*)(ws + 14680064);
  _Float16* valsh = (_Float16*)(ws + 18874368);

  f32_to_f16_k<<<8192, 256, 0, stream>>>(x, xh, 2 * 2048 * 512);
  f32_to_f16_k<<<3072, 256, 0, stream>>>(w_qkv, wqkvh, 1536 * 512);
  f32_to_f16_k<<<1024, 256, 0, stream>>>(w_o, woh, 512 * 512);

  gemm_qkv_k<<<768, 256, 0, stream>>>(xh, wqkvh, b_qkv, qb, kb, vT);
  attn_k<<<256, 256, 0, stream>>>(qb, kb, vT, pmask, valsh);
  gemm_out_k<<<256, 256, 0, stream>>>(valsh, woh, b_o, out);
}
